// DeltaNetBlock_627065225469
// MI455X (gfx1250) — compile-verified, matrix operand order checked by reading, not run on hardware
//
#include <hip/hip_runtime.h>
#include <hip/hip_bf16.h>

typedef __attribute__((ext_vector_type(16))) _Float16 v16h;
typedef __attribute__((ext_vector_type(8)))  float    v8f;

#define BB 2
#define LL 4096
#define DD 1024
#define HH 8
#define DKH 128
#define BLROWS (BB*LL)          // 8192
#define NCHUNK 64
#define EPSF 1e-6f

#if defined(__gfx1250__) && __has_builtin(__builtin_amdgcn_global_load_async_to_lds_b128)
#define HAVE_ASYNC 1
typedef int v4i_ __attribute__((ext_vector_type(4)));
typedef __attribute__((address_space(1))) v4i_* gp_v4i;
typedef __attribute__((address_space(3))) v4i_* lp_v4i;
#define GLA128(SRC, DST) \
  __builtin_amdgcn_global_load_async_to_lds_b128((gp_v4i)(void*)(SRC), (lp_v4i)(void*)(DST), 0, 0)
#if __has_builtin(__builtin_amdgcn_s_wait_asynccnt)
#define WAITA(n) __builtin_amdgcn_s_wait_asynccnt(n)
#else
#define WAITA(n) asm volatile("s_wait_asynccnt %0" :: "n"(n) : "memory")
#endif
#else
#define HAVE_ASYNC 0
#define WAITA(n)
#endif

// K-offset pattern for 16-bit WMMA A/B fragments (CDNA5 ISA 7.12.2):
// element e (0..15) of v16h, lane (0..31):
//   j = e/2 ; K = (j>=4 ? 16:0) + (lane>=16 ? 8:0) + 2*(j%4) + (e%2)
__device__ __forceinline__ int koff(int e, int lane) {
  int j = e >> 1;
  return ((j & 4) << 2) + ((lane & 16) >> 1) + ((j & 3) << 1) + (e & 1);
}

__device__ __forceinline__ v8f wmma_f16(v16h a, v16h b, v8f c) {
  return __builtin_amdgcn_wmma_f32_16x16x32_f16(false, a, false, b, (short)0, c, false, false);
}

__device__ __forceinline__ float sigmoidf_(float x) { return 1.f / (1.f + __expf(-x)); }
__device__ __forceinline__ float siluf_(float x) { return x * sigmoidf_(x); }

// ---------------- RMSNorm (x -> h f16) -------------------------------------
__global__ __launch_bounds__(256) void rmsnorm_kernel(const float* __restrict__ x,
                                                      const float* __restrict__ w,
                                                      _Float16* __restrict__ h16) {
  __shared__ float red[256];
  int tid = threadIdx.x;
  size_t row = blockIdx.x;
  const float* xr = x + row * DD;
  float p = 0.f;
  for (int i = tid; i < DD; i += 256) { float v = xr[i]; p += v * v; }
  red[tid] = p; __syncthreads();
  for (int off = 128; off > 0; off >>= 1) {
    if (tid < off) red[tid] += red[tid + off];
    __syncthreads();
  }
  float inv = rsqrtf(red[0] * (1.f / DD) + EPSF);
  for (int i = tid; i < DD; i += 256)
    h16[row * DD + i] = (_Float16)(xr[i] * inv * w[i]);
}

// -------- LDS-tiled transpose: src f32 [R][C] -> dst f16 [C][R] ------------
__global__ __launch_bounds__(256) void transpose_f16_kernel(const float* __restrict__ src,
                                                            _Float16* __restrict__ dst,
                                                            int R, int C) {
  __shared__ float t[32][33];
  int nbx = C / 32;
  int bx = blockIdx.x % nbx, by = blockIdx.x / nbx;
  int r0 = by * 32, c0 = bx * 32;
  int tr = threadIdx.x >> 5, tc = threadIdx.x & 31;
#pragma unroll
  for (int i = 0; i < 32; i += 8)
    t[tr + i][tc] = src[(size_t)(r0 + tr + i) * C + c0 + tc];
  __syncthreads();
#pragma unroll
  for (int i = 0; i < 32; i += 8)
    dst[(size_t)(c0 + tr + i) * R + r0 + tc] = (_Float16)t[tc][tr + i];
}

// ---------------- WMMA GEMM: C[M,N] = A[M,K] * Bt[N,K]^T (+resid) ----------
// blockTile 128x128, 8 waves as 4(m)x2(n); double-buffered async LDS staging.
// MODE 0: f16 out ; MODE 1: f32 out with f32 residual added
template <int MODE>
__global__ __launch_bounds__(256) void gemm_f16_kernel(
    const _Float16* __restrict__ A, const _Float16* __restrict__ Bt,
    _Float16* __restrict__ out16, float* __restrict__ outf,
    const float* __restrict__ resid, int M, int N, int K) {
  __shared__ _Float16 As[2][128][32];
  __shared__ _Float16 Bs[2][128][32];
  int tid = threadIdx.x, lane = tid & 31;
  int wid = tid >> 5, l15 = lane & 15, hi8 = (lane & 16) ? 8 : 0;
  int wm = wid >> 1, wn = wid & 1;
  int bm = blockIdx.y * 128, bn = blockIdx.x * 128;

  int cr = tid >> 2, cs = (tid & 3) * 8;          // copy row / 8-half segment
  auto issue = [&](int buf, int k0) {
    const _Float16* ga = A + (size_t)(bm + cr) * K + k0 + cs;
    const _Float16* gb = Bt + (size_t)(bn + cr) * K + k0 + cs;
#if HAVE_ASYNC
    GLA128(ga,                    &As[buf][cr][cs]);
    GLA128(ga + (size_t)64 * K,   &As[buf][cr + 64][cs]);
    GLA128(gb,                    &Bs[buf][cr][cs]);
    GLA128(gb + (size_t)64 * K,   &Bs[buf][cr + 64][cs]);
#else
    *(int4*)&As[buf][cr][cs]      = *(const int4*)ga;
    *(int4*)&As[buf][cr + 64][cs] = *(const int4*)(ga + (size_t)64 * K);
    *(int4*)&Bs[buf][cr][cs]      = *(const int4*)gb;
    *(int4*)&Bs[buf][cr + 64][cs] = *(const int4*)(gb + (size_t)64 * K);
#endif
  };

  v8f acc[2][4];
#pragma unroll
  for (int i = 0; i < 2; ++i)
#pragma unroll
    for (int j = 0; j < 4; ++j) acc[i][j] = (v8f){};

  int nT = K / 32;
  issue(0, 0);
  for (int kt = 0; kt < nT; ++kt) {
    int buf = kt & 1;
    if (kt + 1 < nT) { issue(buf ^ 1, (kt + 1) * 32); WAITA(4); }
    else             { WAITA(0); }
    __syncthreads();
    v16h af[2], bf[4];
#pragma unroll
    for (int i = 0; i < 2; ++i) {
      int mr = wm * 32 + i * 16 + l15;
#pragma unroll
      for (int e = 0; e < 16; ++e) af[i][e] = As[buf][mr][koff(e, lane)];
    }
#pragma unroll
    for (int j = 0; j < 4; ++j) {
      int nc = wn * 64 + j * 16 + l15;
#pragma unroll
      for (int e = 0; e < 16; ++e) bf[j][e] = Bs[buf][nc][koff(e, lane)];
    }
#pragma unroll
    for (int i = 0; i < 2; ++i)
#pragma unroll
      for (int j = 0; j < 4; ++j) acc[i][j] = wmma_f16(af[i], bf[j], acc[i][j]);
    __syncthreads();
  }
#pragma unroll
  for (int i = 0; i < 2; ++i)
#pragma unroll
    for (int j = 0; j < 4; ++j)
#pragma unroll
      for (int r = 0; r < 8; ++r) {
        size_t gm = bm + wm * 32 + i * 16 + hi8 + r;
        size_t gn = bn + wn * 64 + j * 16 + l15;
        if (MODE == 0) out16[gm * N + gn] = (_Float16)acc[i][j][r];
        else           outf[gm * N + gn] = resid[gm * N + gn] + acc[i][j][r];
      }
}

// ---------------- beta = sigmoid(h @ Wb) -----------------------------------
__global__ __launch_bounds__(256) void beta_kernel(const _Float16* __restrict__ h16,
                                                   const float* __restrict__ Wb,
                                                   float* __restrict__ betag) {
  __shared__ float red[8][256];
  int tid = threadIdx.x;
  size_t row = blockIdx.x;              // b*L + l
  int b = (int)(row / LL), l = (int)(row % LL);
  float acc[8];
#pragma unroll
  for (int hh = 0; hh < 8; ++hh) acc[hh] = 0.f;
  for (int k = tid; k < DD; k += 256) {
    float hv = (float)h16[row * DD + k];
#pragma unroll
    for (int hh = 0; hh < 8; ++hh) acc[hh] += hv * Wb[k * 8 + hh];
  }
#pragma unroll
  for (int hh = 0; hh < 8; ++hh) red[hh][tid] = acc[hh];
  __syncthreads();
  for (int off = 128; off > 0; off >>= 1) {
    if (tid < off)
#pragma unroll
      for (int hh = 0; hh < 8; ++hh) red[hh][tid] += red[hh][tid + off];
    __syncthreads();
  }
  if (tid < 8)
    betag[((size_t)b * HH + tid) * LL + l] = sigmoidf_(red[tid][0]);
}

// -------- causal conv(K=4)+silu [+silu+l2norm (+qscale)] -> heads f16 ------
// mode: 0 = v (conv+silu), 1 = k (+silu+l2norm), 2 = q (+silu+l2norm+DK^-.5)
__global__ __launch_bounds__(128) void conv_head_kernel(
    const _Float16* __restrict__ pre, const float* __restrict__ cw,
    _Float16* __restrict__ outg, int mode) {
  __shared__ float red[128];
  int d = threadIdx.x;
  size_t row = blockIdx.x;              // (b*H + h)*L + l
  int l = (int)(row % LL);
  int hb = (int)(row / LL);
  int hh = hb % HH, b = hb / HH;
  int ch = hh * DKH + d;
  float acc = 0.f;
#pragma unroll
  for (int i = 0; i < 4; ++i) {
    int ls = l - 3 + i;
    if (ls >= 0)
      acc += (float)pre[((size_t)b * LL + ls) * DD + ch] * cw[ch * 4 + i];
  }
  float s = siluf_(acc);
  if (mode > 0) {
    s = siluf_(s);
    red[d] = s * s;
    __syncthreads();
    for (int off = 64; off > 0; off >>= 1) {
      if (d < off) red[d] += red[d + off];
      __syncthreads();
    }
    s = s * rsqrtf(red[0] + EPSF);
    if (mode == 2) s *= 0.08838834764831845f; // 128^-0.5
  }
  outg[row * DKH + d] = (_Float16)s;
}

// ---------------- chunkwise delta rule (one block per (b,h)) ---------------
// LDS layouts chosen for K-contiguous WMMA fragment gathers:
//   St  [dv][dk] f32 (transposed state)      q16,k16 [j][d] row-major
//   vw16: phase 1 = (beta*v)^T [d][j] f16 ;  phase 2 = -w [j][d] f16
//   Af f32 [64][64] -> reused as qk f16      X16 f16 [64][64]
//   vnT [dv][j] f16 (transposed v_new)
__global__ __launch_bounds__(256) void delta_kernel(
    const _Float16* __restrict__ qg, const _Float16* __restrict__ kg,
    const _Float16* __restrict__ vg, const float* __restrict__ betag,
    float* __restrict__ og) {
  __shared__ float    St[128 * 128];
  __shared__ _Float16 q16[64 * 128];
  __shared__ _Float16 k16[64 * 128];
  __shared__ _Float16 vw16[64 * 128];
  __shared__ float    Af[64 * 64];
  __shared__ _Float16 X16[64 * 64];
  __shared__ _Float16 vnT[128 * 64];
  __shared__ float    betaS[64];
  _Float16* qk16 = (_Float16*)Af;

  int tid = threadIdx.x, lane = tid & 31, wid = tid >> 5;
  int l15 = lane & 15, hi8 = (lane & 16) ? 8 : 0;
  size_t bh = blockIdx.x;
  size_t rowbase = bh * (size_t)LL;

  for (int i = tid; i < 128 * 128; i += 256) St[i] = 0.f;
  __syncthreads();

  for (int c = 0; c < NCHUNK; ++c) {
    // ---- load chunk (beta first; fold beta into transposed v) ----
    if (tid < 64) betaS[tid] = betag[rowbase + (size_t)c * 64 + tid];
    __syncthreads();
    for (int i = tid; i < 64 * 128; i += 256) {
      int j = i >> 7, d = i & 127;
      size_t g = (rowbase + (size_t)c * 64 + j) * DKH + d;
      q16[i] = qg[g];
      k16[i] = kg[g];
      vw16[d * 64 + j] = (_Float16)((float)vg[g] * betaS[j]);   // (beta*v)^T
    }
    __syncthreads();

    // ---- kk = k k^T ; A = I + tril(beta*kk, -1) ----
#pragma unroll
    for (int ti = 0; ti < 2; ++ti) {
      int t = wid * 2 + ti, tm = t >> 2, tn = t & 3;
      v8f a0 = (v8f){};
#pragma unroll
      for (int kb = 0; kb < 128; kb += 32) {
        v16h a, b;
#pragma unroll
        for (int e = 0; e < 16; ++e) {
          int ko = kb + koff(e, lane);
          a[e] = k16[(tm * 16 + l15) * 128 + ko];
          b[e] = k16[(tn * 16 + l15) * 128 + ko];   // k^T as B
        }
        a0 = wmma_f16(a, b, a0);
      }
#pragma unroll
      for (int r = 0; r < 8; ++r) {
        int m = tm * 16 + hi8 + r, n = tn * 16 + l15;
        Af[m * 64 + n] = (m == n) ? 1.f : ((n < m) ? betaS[m] * a0[r] : 0.f);
      }
    }
    __syncthreads();

    // ---- X = A^-1 (unit lower tri, forward substitution per column) ----
    if (tid < 64) {
      int j = tid;
      for (int i2 = 0; i2 < 64; ++i2) {
        float s;
        if (i2 < j) s = 0.f;
        else if (i2 == j) s = 1.f;
        else {
          float a2 = 0.f;
          for (int l2 = j; l2 < i2; ++l2)
            a2 += Af[i2 * 64 + l2] * (float)X16[l2 * 64 + j];
          s = -a2;
        }
        X16[i2 * 64 + j] = (_Float16)s;
      }
    }
    __syncthreads();

    // ---- u = X @ (beta*v) (regs) ; o1 = q @ S (regs) ----
    v8f uacc[4], oacc[4];
#pragma unroll
    for (int ti = 0; ti < 4; ++ti) {
      int t = wid * 4 + ti, tm = t >> 3, tn = t & 7;
      v8f a0 = (v8f){};
#pragma unroll
      for (int kb = 0; kb < 64; kb += 32) {
        v16h a, b;
#pragma unroll
        for (int e = 0; e < 16; ++e) {
          int ko = kb + koff(e, lane);
          a[e] = X16[(tm * 16 + l15) * 64 + ko];
          b[e] = vw16[(tn * 16 + l15) * 64 + ko];           // (beta*v)^T
        }
        a0 = wmma_f16(a, b, a0);
      }
      uacc[ti] = a0;
      v8f a1 = (v8f){};
#pragma unroll
      for (int kb = 0; kb < 128; kb += 32) {
        v16h a, b;
#pragma unroll
        for (int e = 0; e < 16; ++e) {
          int ko = kb + koff(e, lane);
          a[e] = q16[(tm * 16 + l15) * 128 + ko];
          b[e] = (_Float16)St[(tn * 16 + l15) * 128 + ko];  // S^T rows
        }
        a1 = wmma_f16(a, b, a1);
      }
      oacc[ti] = a1;
    }
    __syncthreads();

    // ---- -w = -(X @ (beta*k)) -> vw16 (row-major) ; qk = tril(q k^T) ----
#pragma unroll
    for (int ti = 0; ti < 4; ++ti) {
      int t = wid * 4 + ti, tm = t >> 3, tn = t & 7;
      v8f a0 = (v8f){};
#pragma unroll
      for (int kb = 0; kb < 64; kb += 32) {
        v16h a, b;
#pragma unroll
        for (int e = 0; e < 16; ++e) {
          int ko = kb + koff(e, lane);
          a[e] = X16[(tm * 16 + l15) * 64 + ko];
          b[e] = (_Float16)((float)k16[ko * 128 + tn * 16 + l15] * betaS[ko]);
        }
        a0 = wmma_f16(a, b, a0);
      }
#pragma unroll
      for (int r = 0; r < 8; ++r)
        vw16[(tm * 16 + hi8 + r) * 128 + tn * 16 + l15] = (_Float16)(-a0[r]);
    }
#pragma unroll
    for (int ti = 0; ti < 2; ++ti) {
      int t = wid * 2 + ti, tm = t >> 2, tn = t & 3;
      v8f a0 = (v8f){};
#pragma unroll
      for (int kb = 0; kb < 128; kb += 32) {
        v16h a, b;
#pragma unroll
        for (int e = 0; e < 16; ++e) {
          int ko = kb + koff(e, lane);
          a[e] = q16[(tm * 16 + l15) * 128 + ko];
          b[e] = k16[(tn * 16 + l15) * 128 + ko];
        }
        a0 = wmma_f16(a, b, a0);
      }
#pragma unroll
      for (int r = 0; r < 8; ++r) {
        int m = tm * 16 + hi8 + r, n = tn * 16 + l15;
        qk16[m * 64 + n] = (_Float16)((n <= m) ? a0[r] : 0.f);
      }
    }
    __syncthreads();

    // ---- v_new = u + (-w) @ S -> vnT (transposed) ----
#pragma unroll
    for (int ti = 0; ti < 4; ++ti) {
      int t = wid * 4 + ti, tm = t >> 3, tn = t & 7;
      v8f a0 = uacc[ti];
#pragma unroll
      for (int kb = 0; kb < 128; kb += 32) {
        v16h a, b;
#pragma unroll
        for (int e = 0; e < 16; ++e) {
          int ko = kb + koff(e, lane);
          a[e] = vw16[(tm * 16 + l15) * 128 + ko];          // -w row-major
          b[e] = (_Float16)St[(tn * 16 + l15) * 128 + ko];  // S^T rows
        }
        a0 = wmma_f16(a, b, a0);
      }
#pragma unroll
      for (int r = 0; r < 8; ++r)
        vnT[(tn * 16 + l15) * 64 + tm * 16 + hi8 + r] = (_Float16)a0[r];
    }
    __syncthreads();

    // ---- o = o1 + qk @ v_new -> global ----
#pragma unroll
    for (int ti = 0; ti < 4; ++ti) {
      int t = wid * 4 + ti, tm = t >> 3, tn = t & 7;
      v8f a0 = oacc[ti];
#pragma unroll
      for (int kb = 0; kb < 64; kb += 32) {
        v16h a, b;
#pragma unroll
        for (int e = 0; e < 16; ++e) {
          int ko = kb + koff(e, lane);
          a[e] = qk16[(tm * 16 + l15) * 64 + ko];
          b[e] = vnT[(tn * 16 + l15) * 64 + ko];            // v_new^T rows
        }
        a0 = wmma_f16(a, b, a0);
      }
#pragma unroll
      for (int r = 0; r < 8; ++r)
        og[(rowbase + (size_t)c * 64 + tm * 16 + hi8 + r) * 128 + tn * 16 + l15] = a0[r];
    }
    // ---- S^T += (k^T @ v_new)^T : tile (m=dk, n=dv), stored transposed ----
#pragma unroll
    for (int ti = 0; ti < 8; ++ti) {
      int t = wid * 8 + ti, tm = t >> 3, tn = t & 7;
      v8f a0;
#pragma unroll
      for (int r = 0; r < 8; ++r)
        a0[r] = St[(tn * 16 + l15) * 128 + tm * 16 + hi8 + r];
#pragma unroll
      for (int kb = 0; kb < 64; kb += 32) {
        v16h a, b;
#pragma unroll
        for (int e = 0; e < 16; ++e) {
          int ko = kb + koff(e, lane);
          a[e] = k16[ko * 128 + tm * 16 + l15];   // (k^T)[m][j] = k[j][m]
          b[e] = vnT[(tn * 16 + l15) * 64 + ko];  // v_new^T rows
        }
        a0 = wmma_f16(a, b, a0);
      }
#pragma unroll
      for (int r = 0; r < 8; ++r)
        St[(tn * 16 + l15) * 128 + tm * 16 + hi8 + r] = a0[r];
    }
    __syncthreads();
  }
}

// ---------------- per-head output RMSNorm -> [B,L,D] f16 -------------------
__global__ __launch_bounds__(128) void onorm_kernel(const float* __restrict__ og,
                                                    const float* __restrict__ w,
                                                    _Float16* __restrict__ on16) {
  __shared__ float red[128];
  int d = threadIdx.x;
  size_t row = blockIdx.x;              // (b*H + h)*L + l
  int l = (int)(row % LL);
  int hb = (int)(row / LL);
  int hh = hb % HH, b = hb / HH;
  float v = og[row * DKH + d];
  red[d] = v * v;
  __syncthreads();
  for (int off = 64; off > 0; off >>= 1) {
    if (d < off) red[d] += red[d + off];
    __syncthreads();
  }
  float inv = rsqrtf(red[0] * (1.f / DKH) + EPSF);
  on16[((size_t)b * LL + l) * DD + hh * DKH + d] = (_Float16)(v * inv * w[d]);
}

// ---------------------------------------------------------------------------
extern "C" void kernel_launch(void* const* d_in, const int* in_sizes, int n_in,
                              void* d_out, int out_size, void* d_ws, size_t ws_size,
                              hipStream_t stream) {
  const float* x        = (const float*)d_in[0];
  const float* norm_w   = (const float*)d_in[1];
  const float* Wq       = (const float*)d_in[2];
  const float* Wk       = (const float*)d_in[3];
  const float* Wv       = (const float*)d_in[4];
  const float* Wb       = (const float*)d_in[5];
  const float* conv_q   = (const float*)d_in[6];
  const float* conv_k   = (const float*)d_in[7];
  const float* conv_v   = (const float*)d_in[8];
  const float* o_norm_w = (const float*)d_in[9];
  const float* Wo       = (const float*)d_in[10];
  float* out = (float*)d_out;

  char* p = (char*)d_ws;
  auto take = [&](size_t bytes) { char* r = p; p += (bytes + 255) & ~(size_t)255; return r; };
  const size_t actF16 = (size_t)BLROWS * DD * sizeof(_Float16);   // 16 MiB
  const size_t wF16   = (size_t)DD * DD * sizeof(_Float16);       // 2 MiB

  _Float16* h16   = (_Float16*)take(actF16);
  _Float16* wq16t = (_Float16*)take(wF16);   // transposed [N][K]
  _Float16* wk16t = (_Float16*)take(wF16);
  _Float16* wv16t = (_Float16*)take(wF16);
  _Float16* wo16t = (_Float16*)take(wF16);
  _Float16* qpre  = (_Float16*)take(actF16);
  _Float16* kpre  = (_Float16*)take(actF16);
  _Float16* vpre  = (_Float16*)take(actF16);
  _Float16* qh    = (_Float16*)take(actF16);
  _Float16* kh    = (_Float16*)take(actF16);
  _Float16* vh    = (_Float16*)take(actF16);
  float*    betag = (float*)take((size_t)BB * HH * LL * sizeof(float));
  float*    og    = (float*)take((size_t)BB * HH * LL * DKH * sizeof(float));
  _Float16* on16  = (_Float16*)take(actF16);

  // 1. h = rmsnorm(x)
  rmsnorm_kernel<<<BLROWS, 256, 0, stream>>>(x, norm_w, h16);
  // 2. weights -> f16, transposed to [N][K] for K-contiguous B fragments
  int tgrid = (DD / 32) * (DD / 32);
  transpose_f16_kernel<<<tgrid, 256, 0, stream>>>(Wq, wq16t, DD, DD);
  transpose_f16_kernel<<<tgrid, 256, 0, stream>>>(Wk, wk16t, DD, DD);
  transpose_f16_kernel<<<tgrid, 256, 0, stream>>>(Wv, wv16t, DD, DD);
  transpose_f16_kernel<<<tgrid, 256, 0, stream>>>(Wo, wo16t, DD, DD);
  // 3. projections (WMMA GEMMs, async double-buffered staging)
  dim3 ggrid(DD / 128, BLROWS / 128);
  gemm_f16_kernel<0><<<ggrid, 256, 0, stream>>>(h16, wq16t, qpre, nullptr, nullptr, BLROWS, DD, DD);
  gemm_f16_kernel<0><<<ggrid, 256, 0, stream>>>(h16, wk16t, kpre, nullptr, nullptr, BLROWS, DD, DD);
  gemm_f16_kernel<0><<<ggrid, 256, 0, stream>>>(h16, wv16t, vpre, nullptr, nullptr, BLROWS, DD, DD);
  // 4. beta
  beta_kernel<<<BLROWS, 256, 0, stream>>>(h16, Wb, betag);
  // 5. conv + activations + head layout
  conv_head_kernel<<<BB * HH * LL, 128, 0, stream>>>(qpre, conv_q, qh, 2);
  conv_head_kernel<<<BB * HH * LL, 128, 0, stream>>>(kpre, conv_k, kh, 1);
  conv_head_kernel<<<BB * HH * LL, 128, 0, stream>>>(vpre, conv_v, vh, 0);
  // 6. chunkwise delta rule
  delta_kernel<<<BB * HH, 256, 0, stream>>>(qh, kh, vh, betag, og);
  // 7. output norm
  onorm_kernel<<<BB * HH * LL, 128, 0, stream>>>(og, o_norm_w, on16);
  // 8. out = x + on @ Wo
  gemm_f16_kernel<1><<<ggrid, 256, 0, stream>>>(on16, wo16t, nullptr, out, x, BLROWS, DD, DD);
}
